// LayerBuilder_40673340293650
// MI455X (gfx1250) — matrix operand order repaired and checked by reading, not compiled, not run on hardware
//
#include <hip/hip_runtime.h>

// PointNet++ MSG set-abstraction for MI455X (gfx1250, wave32, WMMA).
// - All matmuls (3-layer MLP per scale + aggregation) via v_wmma_f32_16x16x32_f16
//   (f32 accumulate), activations staged in LDS as f16 with bank-conflict-padded,
//   16B-aligned row strides.
// - Weights / point features pre-packed to f16 once (layer-1 weights zero-padded
//   to 64 rows and row-permuted so gathered channels are [points|relxyz|pad]).
// - Gathered point-features and the aggregation feature tile are moved with
//   GLOBAL_LOAD_ASYNC_TO_LDS_B128 (ASYNCcnt) -- global -> LDS without VGPRs.

typedef __attribute__((ext_vector_type(16))) _Float16 v16h;
typedef __attribute__((ext_vector_type(8)))  float    v8f;

static constexpr int BATCH = 2;
static constexpr int NPTS  = 16384;
static constexpr int M     = 4096;
static constexpr int CINF  = 32;

// d_out layout (floats): new_xyz [2,4096,3] | new_points [2,4096,64] | fps_idx (int bits) [2,4096]
static constexpr int OUT_NEWPTS = BATCH * M * 3;               // 24576
static constexpr int OUT_FPS    = OUT_NEWPTS + BATCH * M * 64; // 548864

// Workspace layout. f16 weight region first (padded/permuted layer-1 rows),
// then ball-query index lists, f16 concat features, f16 points.
static constexpr size_t WOFF[10] = {      // offsets in halves within weight region
    0,      2048,  3072,            // s0: 64x32 | 32x32 | 32x64
    5120,   9216,  13312,           // s1: 64x64 | 64x64 | 64x128
    21504,  25600, 31744,           // s2: 64x64 | 64x96 | 96x128
    44032                           // agg: 320x64
};
static constexpr size_t W_HALVES   = 64512;                     // total halves
static constexpr size_t WS_IDX0    = W_HALVES * 2;              // 129024
static constexpr size_t WS_IDX1    = WS_IDX0 + 8192ull * 16 * 4;
static constexpr size_t WS_IDX2    = WS_IDX1 + 8192ull * 32 * 4;
static constexpr size_t WS_FEAT    = WS_IDX2 + 8192ull * 64 * 4;
static constexpr size_t WS_PTS16   = WS_FEAT + 8192ull * 320 * 2;

// ---------------------------------------------------------------------------
// f32 -> f16 conversion with optional zero row padding (dstRows >= rows).
// ---------------------------------------------------------------------------
__global__ __launch_bounds__(256)
void cvt_f16_kernel(const float* __restrict__ src, _Float16* __restrict__ dst,
                    int rows, int cols, int dstRows)
{
  const int e = blockIdx.x * 256 + threadIdx.x;
  if (e >= dstRows * cols) return;
  const int r = e / cols;
  dst[e] = (r < rows) ? (_Float16)src[e] : (_Float16)0.0f;
}

// ---------------------------------------------------------------------------
// Layer-1 weight pre-pack: f16, 64 rows, rows permuted to match the gathered
// channel order [points(32) | relxyz(3) | zero pad]:
//   dst row d<32 -> src row 3+d (point-feature rows), 32<=d<35 -> src row d-32
//   (xyz rows), else zeros. src is [35 x cols].
// ---------------------------------------------------------------------------
__global__ __launch_bounds__(256)
void cvt_w1_kernel(const float* __restrict__ src, _Float16* __restrict__ dst,
                   int cols)
{
  const int e = blockIdx.x * 256 + threadIdx.x;
  if (e >= 64 * cols) return;
  const int r = e / cols, c = e % cols;
  const int sr = (r < 32) ? (3 + r) : (r < 35 ? r - 32 : -1);
  dst[e] = (sr >= 0) ? (_Float16)src[sr * cols + c] : (_Float16)0.0f;
}

// ---------------------------------------------------------------------------
// Kernel 1: farthest point sampling (+ new_xyz gather). grid = B, block = 1024.
// ---------------------------------------------------------------------------
__global__ __launch_bounds__(1024, 1)
void fps_kernel(const float* __restrict__ xyz, float* __restrict__ newxyz,
                int* __restrict__ fpsIdx)
{
  const int b = blockIdx.x;
  const int tid = threadIdx.x;
  const float* xb = xyz + (size_t)b * NPTS * 3;

  float px[16], py[16], pz[16], dist[16];
#pragma unroll
  for (int i = 0; i < 16; ++i) {
    const int n = tid + i * 1024;
    px[i] = xb[n * 3 + 0]; py[i] = xb[n * 3 + 1]; pz[i] = xb[n * 3 + 2];
    dist[i] = 1e10f;
  }

  __shared__ float s_rv[32];
  __shared__ int   s_ri[32];
  __shared__ float s_px, s_py, s_pz;

  if (tid == 0) {
    s_px = xb[0]; s_py = xb[1]; s_pz = xb[2];
    fpsIdx[b * M + 0] = 0;
    newxyz[(b * M + 0) * 3 + 0] = xb[0];
    newxyz[(b * M + 0) * 3 + 1] = xb[1];
    newxyz[(b * M + 0) * 3 + 2] = xb[2];
  }
  __syncthreads();

  const int lane = tid & 31, wave = tid >> 5;
  for (int j = 1; j < M; ++j) {
    const float lx = s_px, ly = s_py, lz = s_pz;
    float bd = -1.0f; int bi = 0;
#pragma unroll
    for (int i = 0; i < 16; ++i) {
      const float dx = px[i] - lx, dy = py[i] - ly, dz = pz[i] - lz;
      float d = dx * dx + dy * dy + dz * dz;
      d = fminf(dist[i], d);
      dist[i] = d;
      const int n = tid + i * 1024;
      if (d > bd || (d == bd && n < bi)) { bd = d; bi = n; }
    }
#pragma unroll
    for (int off = 16; off > 0; off >>= 1) {
      const float od = __shfl_down(bd, off, 32);
      const int   oi = __shfl_down(bi, off, 32);
      if (od > bd || (od == bd && oi < bi)) { bd = od; bi = oi; }
    }
    if (lane == 0) { s_rv[wave] = bd; s_ri[wave] = bi; }
    __syncthreads();
    if (wave == 0) {
      bd = s_rv[lane]; bi = s_ri[lane];
#pragma unroll
      for (int off = 16; off > 0; off >>= 1) {
        const float od = __shfl_down(bd, off, 32);
        const int   oi = __shfl_down(bi, off, 32);
        if (od > bd || (od == bd && oi < bi)) { bd = od; bi = oi; }
      }
      if (lane == 0) {
        const float nx = xb[bi * 3 + 0], ny = xb[bi * 3 + 1], nz = xb[bi * 3 + 2];
        s_px = nx; s_py = ny; s_pz = nz;
        fpsIdx[b * M + j] = bi;
        newxyz[(b * M + j) * 3 + 0] = nx;
        newxyz[(b * M + j) * 3 + 1] = ny;
        newxyz[(b * M + j) * 3 + 2] = nz;
      }
    }
    __syncthreads();
  }
}

// ---------------------------------------------------------------------------
// Kernel 2: ball query, all 3 radii in one ordered pass. One wave per query;
// wave32 ballot/popc compaction preserves index order; pads misses with the
// first hit (or N-1 when there is no hit), matching the reference.
// ---------------------------------------------------------------------------
__global__ __launch_bounds__(256)
void ballquery_kernel(const float* __restrict__ xyz,
                      const float* __restrict__ newxyz,
                      int* __restrict__ idx0, int* __restrict__ idx1,
                      int* __restrict__ idx2)
{
  const int lane = threadIdx.x & 31;
  const int q = blockIdx.x * 8 + (threadIdx.x >> 5);   // 0 .. B*M-1
  const int b = q / M;
  const float* xb = xyz + (size_t)b * NPTS * 3;
  const float cx = newxyz[q * 3 + 0], cy = newxyz[q * 3 + 1], cz = newxyz[q * 3 + 2];
  int* o0 = idx0 + (size_t)q * 16;
  int* o1 = idx1 + (size_t)q * 32;
  int* o2 = idx2 + (size_t)q * 64;
  const float r0 = 0.04f, r1 = 0.16f, r2 = 0.64f;      // radii^2
  int c0 = 0, c1 = 0, c2 = 0;
  int f0 = NPTS - 1, f1 = NPTS - 1, f2 = NPTS - 1;
  const unsigned lt = (1u << lane) - 1u;

  for (int n0 = 0; n0 < NPTS; n0 += 32) {
    const int n = n0 + lane;
    const float dx = xb[n * 3 + 0] - cx;
    const float dy = xb[n * 3 + 1] - cy;
    const float dz = xb[n * 3 + 2] - cz;
    const float d2 = dx * dx + dy * dy + dz * dz;
    const unsigned m2 = (unsigned)__ballot(d2 < r2);
    const unsigned m1 = (unsigned)__ballot(d2 < r1);
    const unsigned m0 = (unsigned)__ballot(d2 < r0);
    if (c0 == 0 && m0) f0 = n0 + __ffs((int)m0) - 1;
    if (c1 == 0 && m1) f1 = n0 + __ffs((int)m1) - 1;
    if (c2 == 0 && m2) f2 = n0 + __ffs((int)m2) - 1;
    if (m0 & (1u << lane)) { const int p = c0 + __popc(m0 & lt); if (p < 16) o0[p] = n; }
    if (m1 & (1u << lane)) { const int p = c1 + __popc(m1 & lt); if (p < 32) o1[p] = n; }
    if (m2 & (1u << lane)) { const int p = c2 + __popc(m2 & lt); if (p < 64) o2[p] = n; }
    c0 += __popc(m0); c1 += __popc(m1); c2 += __popc(m2);
    if (c0 >= 16 && c1 >= 32 && c2 >= 64) break;
  }
  const int s0 = c0 < 16 ? c0 : 16;
  const int s1 = c1 < 32 ? c1 : 32;
  const int s2 = c2 < 64 ? c2 : 64;
  for (int s = s0 + lane; s < 16; s += 32) o0[s] = f0;
  for (int s = s1 + lane; s < 32; s += 32) o1[s] = f1;
  for (int s = s2 + lane; s < 64; s += 32) o2[s] = f2;
}

// ---------------------------------------------------------------------------
// Async global -> LDS 16-byte copy (ASYNCcnt path), per-lane addresses.
// ---------------------------------------------------------------------------
__device__ __forceinline__ void async_copy_b128(const void* gsrc, void* ldsDst)
{
  const unsigned lds = (unsigned)(size_t)ldsDst;   // low 32 bits = LDS address
  asm volatile("global_load_async_to_lds_b128 %0, %1, off"
               :: "v"(lds), "v"(gsrc) : "memory");
}
__device__ __forceinline__ void async_wait0()
{
  asm volatile("s_wait_asynccnt 0x0" ::: "memory");
}

// ---------------------------------------------------------------------------
// WMMA MLP layer: 16-row M-tile per wave. A from LDS (CDNA5 16-bit A layout:
// lane=row, K halves split 0-7/8-15 across lane halves), B direct f16 loads
// from the pre-packed weight region (lane=N, 16 K halves), D written back to
// LDS as f16 with bias + optional relu fused.
// ---------------------------------------------------------------------------
__device__ __forceinline__ void mlp_layer(const _Float16* bin, int sin,
                                          _Float16* bout, int sout,
                                          const _Float16* __restrict__ W,
                                          const float* __restrict__ bias,
                                          int ksteps, int ntiles, int cout,
                                          int rowBase, int lane, bool relu)
{
  const int arow  = rowBase + (lane & 15);
  const int kbase = (lane & 16) ? 8 : 0;
  for (int nt = 0; nt < ntiles; ++nt) {
    v8f acc = {0.f, 0.f, 0.f, 0.f, 0.f, 0.f, 0.f, 0.f};
    const float bval = bias[nt * 16 + (lane & 15)];
    for (int ks = 0; ks < ksteps; ++ks) {
      union { v16h h; uint4 u[2]; } A;
      const _Float16* ap = bin + arow * sin + ks * 32 + kbase;
      A.u[0] = *reinterpret_cast<const uint4*>(ap);       // K kbase..kbase+7
      A.u[1] = *reinterpret_cast<const uint4*>(ap + 16);  // K kbase+16..+23
      union { v16h h; uint4 u[2]; } B;
      const _Float16* wr = W + (size_t)(ks * 32 + kbase) * cout + nt * 16 + (lane & 15);
#pragma unroll
      for (int s = 0; s < 8; ++s) {
        B.h[s]     = wr[(size_t)s * cout];
        B.h[s + 8] = wr[(size_t)(16 + s) * cout];
      }
      acc = __builtin_amdgcn_wmma_f32_16x16x32_f16(false, A.h, false, B.h,
                                                   (short)0, acc, false, false);
    }
    const int col = nt * 16 + (lane & 15);
    const int r0  = rowBase + ((lane >> 4) << 3);
#pragma unroll
    for (int v = 0; v < 8; ++v) {
      float x = acc[v] + bval;
      if (relu) x = fmaxf(x, 0.0f);
      bout[(r0 + v) * sout + col] = (_Float16)x;
    }
  }
}

// ---------------------------------------------------------------------------
// Kernel 3 (x3 scales): gather -> 3-layer WMMA MLP -> max-pool over K.
// 128 rows (= 128/K query points) per 256-thread WG; each of 8 waves owns
// one 16-row M-tile through the whole chain (LDS ping-pong for layout).
// Gathered channel order (matches permuted W1): [points(32) | relxyz(3) | 0].
// Point features move global->LDS via async b128 (4 chunks / row).
// ---------------------------------------------------------------------------
template<int K, int C1, int C2, int C3, int CHOFF>
__global__ __launch_bounds__(256)
void group_mlp_kernel(const float* __restrict__ xyz,
                      const _Float16* __restrict__ pts16,
                      const float* __restrict__ newxyz,
                      const int* __restrict__ idx,
                      const _Float16* __restrict__ W1, const float* __restrict__ b1,
                      const _Float16* __restrict__ W2, const float* __restrict__ b2,
                      const _Float16* __restrict__ W3, const float* __restrict__ b3,
                      _Float16* __restrict__ featOut)
{
  constexpr int ROWS = 128;
  constexpr int Q    = ROWS / K;
  constexpr int SA   = (C1 > C3 ? C1 : C3) + 8;  // halves; *2 bytes % 16 == 0
  constexpr int SB   = C2 + 8;
  __shared__ alignas(16) _Float16 buf0[ROWS * 72];
  __shared__ alignas(16) _Float16 bufA[ROWS * SA];
  __shared__ alignas(16) _Float16 bufB[ROWS * SB];

  const int tid = threadIdx.x;

  // Prefetch weight streams (global_prefetch_b8) while the gather runs.
  if (tid == 0) {
    __builtin_prefetch(W1, 0, 0);
    __builtin_prefetch(W2, 0, 0);
    __builtin_prefetch(W3, 0, 0);
  }

  // Async gather of point features: cols 0..31, 4x16B chunks per row.
  for (int e = tid; e < ROWS * 4; e += 256) {
    const int r = e >> 2, ch = e & 3;
    const int gq = blockIdx.x * Q + r / K;
    const int kk = r % K;
    const int b  = gq / M;
    const int i  = idx[(size_t)gq * K + kk];
    async_copy_b128(pts16 + ((size_t)b * NPTS + i) * CINF + ch * 8,
                    &buf0[r * 72 + ch * 8]);
  }
  // Rel-xyz (cols 32..34) + zero pad (cols 35..63) via normal DS stores.
  for (int e = tid; e < ROWS * 32; e += 256) {
    const int r = e >> 5, c = (e & 31) + 32;
    const int gq = blockIdx.x * Q + r / K;
    const int kk = r % K;
    const int b  = gq / M;
    const int i  = idx[(size_t)gq * K + kk];
    _Float16 v = (_Float16)0.0f;
    if (c < 35)
      v = (_Float16)(xyz[((size_t)b * NPTS + i) * 3 + (c - 32)] -
                     newxyz[(size_t)gq * 3 + (c - 32)]);
    buf0[r * 72 + c] = v;
  }
  async_wait0();
  __syncthreads();

  const int lane = tid & 31, wave = tid >> 5;
  const int rb = wave * 16;
  mlp_layer(buf0, 72, bufA, SA, W1, b1, 2,       C1 / 16, C1, rb, lane, true);
  mlp_layer(bufA, SA, bufB, SB, W2, b2, C1 / 32, C2 / 16, C2, rb, lane, true);
  mlp_layer(bufB, SB, bufA, SA, W3, b3, C2 / 32, C3 / 16, C3, rb, lane, true);
  __syncthreads();

  // Max-pool over K neighbors (relu output >= 0, so 0 init is safe)
  for (int e = tid; e < Q * C3; e += 256) {
    const int q = e / C3, c = e % C3;
    _Float16 v = (_Float16)0.0f;
    for (int kk = 0; kk < K; ++kk) {
      const _Float16 x = bufA[(q * K + kk) * SA + c];
      v = (x > v) ? x : v;
    }
    const int gq = blockIdx.x * Q + q;
    featOut[(size_t)gq * 320 + CHOFF + c] = v;
  }
}

// ---------------------------------------------------------------------------
// Kernel 4: aggregation GEMM  [8192 x 320] @ [320 x 64] + bias  via WMMA.
// 128 rows per WG; 8 waves x one 16-row tile; K = 10 steps of 32. The f16
// feature tile is staged via async global->LDS b128 copies.
// ---------------------------------------------------------------------------
__global__ __launch_bounds__(256)
void agg_kernel(const _Float16* __restrict__ feat, const _Float16* __restrict__ W,
                const float* __restrict__ bias, float* __restrict__ out)
{
  __shared__ alignas(16) _Float16 buf[128 * 328];
  const int tid = threadIdx.x;
  const int row0 = blockIdx.x * 128;

  if (tid == 0) __builtin_prefetch(W, 0, 0);

  // 128 rows x 320 halves = 128 x 40 16B chunks, async global -> LDS.
  for (int e = tid; e < 128 * 40; e += 256) {
    const int r = e / 40, c = e % 40;
    async_copy_b128(feat + (size_t)(row0 + r) * 320 + c * 8,
                    &buf[r * 328 + c * 8]);
  }
  async_wait0();
  __syncthreads();

  const int lane = tid & 31, wave = tid >> 5;
  const int rb = wave * 16;
  const int arow  = rb + (lane & 15);
  const int kbase = (lane & 16) ? 8 : 0;
  for (int nt = 0; nt < 4; ++nt) {
    v8f acc = {0.f, 0.f, 0.f, 0.f, 0.f, 0.f, 0.f, 0.f};
    const float bval = bias[nt * 16 + (lane & 15)];
    for (int ks = 0; ks < 10; ++ks) {
      union { v16h h; uint4 u[2]; } A;
      const _Float16* ap = buf + arow * 328 + ks * 32 + kbase;
      A.u[0] = *reinterpret_cast<const uint4*>(ap);
      A.u[1] = *reinterpret_cast<const uint4*>(ap + 16);
      union { v16h h; uint4 u[2]; } B;
      const _Float16* wr = W + (size_t)(ks * 32 + kbase) * 64 + nt * 16 + (lane & 15);
#pragma unroll
      for (int s = 0; s < 8; ++s) {
        B.h[s]     = wr[(size_t)s * 64];
        B.h[s + 8] = wr[(size_t)(16 + s) * 64];
      }
      acc = __builtin_amdgcn_wmma_f32_16x16x32_f16(false, A.h, false, B.h,
                                                   (short)0, acc, false, false);
    }
    const int col = nt * 16 + (lane & 15);
    const int r0  = rb + ((lane >> 4) << 3);
#pragma unroll
    for (int v = 0; v < 8; ++v)
      out[(size_t)(row0 + r0 + v) * 64 + col] = acc[v] + bval;
  }
}

// ---------------------------------------------------------------------------
// Host-side launch. d_in order (setup_inputs dict, tuples flattened):
//  0 xyz, 1 points, 2..10 mlp_ws (3 per scale), 11..19 mlp_bs, 20 agg_w, 21 agg_b
// ---------------------------------------------------------------------------
extern "C" void kernel_launch(void* const* d_in, const int* in_sizes, int n_in,
                              void* d_out, int out_size, void* d_ws, size_t ws_size,
                              hipStream_t stream) {
  (void)in_sizes; (void)n_in; (void)out_size; (void)ws_size;
  const float* xyz    = (const float*)d_in[0];
  const float* points = (const float*)d_in[1];
  const float* w[9];  for (int i = 0; i < 9; ++i) w[i]  = (const float*)d_in[2 + i];
  const float* bs[9]; for (int i = 0; i < 9; ++i) bs[i] = (const float*)d_in[11 + i];
  const float* aggw = (const float*)d_in[20];
  const float* aggb = (const float*)d_in[21];

  float* out    = (float*)d_out;
  float* newxyz = out;                    // [2,4096,3]
  float* newpts = out + OUT_NEWPTS;       // [2,4096,64]
  int*   fpsIdx = (int*)(out + OUT_FPS);  // [2,4096] int32 bits

  char* ws = (char*)d_ws;
  _Float16* wf16   = (_Float16*)ws;                 // packed f16 weights
  int*      idx0   = (int*)(ws + WS_IDX0);          // [8192,16]
  int*      idx1   = (int*)(ws + WS_IDX1);          // [8192,32]
  int*      idx2   = (int*)(ws + WS_IDX2);          // [8192,64]
  _Float16* feat   = (_Float16*)(ws + WS_FEAT);     // [8192,320] f16
  _Float16* pts16  = (_Float16*)(ws + WS_PTS16);    // [2,16384,32] f16

  // --- one-time f16 pre-pack ---
  // Layer-1 weights: padded to 64 rows, permuted to [points|xyz|pad] order.
  {
    const int c1[3] = {32, 64, 64};
    const int widx[3] = {0, 3, 6};
    for (int s = 0; s < 3; ++s) {
      const int n = 64 * c1[s];
      cvt_w1_kernel<<<dim3((n + 255) / 256), dim3(256), 0, stream>>>(
          w[widx[s]], wf16 + WOFF[widx[s]], c1[s]);
    }
  }
  // Layers 2/3 weights: straight f16 copies.
  {
    const int li[6]   = {1, 2, 4, 5, 7, 8};
    const int rws[6]  = {32, 32, 64, 64, 64, 96};
    const int cls[6]  = {32, 64, 64, 128, 96, 128};
    for (int t = 0; t < 6; ++t) {
      const int n = rws[t] * cls[t];
      cvt_f16_kernel<<<dim3((n + 255) / 256), dim3(256), 0, stream>>>(
          w[li[t]], wf16 + WOFF[li[t]], rws[t], cls[t], rws[t]);
    }
  }
  cvt_f16_kernel<<<dim3((320 * 64 + 255) / 256), dim3(256), 0, stream>>>(
      aggw, wf16 + WOFF[9], 320, 64, 320);
  cvt_f16_kernel<<<dim3((BATCH * NPTS * CINF + 255) / 256), dim3(256), 0, stream>>>(
      points, pts16, BATCH * NPTS, CINF, BATCH * NPTS);

  fps_kernel<<<dim3(BATCH), dim3(1024), 0, stream>>>(xyz, newxyz, fpsIdx);
  ballquery_kernel<<<dim3(1024), dim3(256), 0, stream>>>(xyz, newxyz, idx0, idx1, idx2);

  // scale 0: K=16, 35->32->32->64,  chan offset 0   (Q=8  -> 1024 blocks)
  group_mlp_kernel<16, 32, 32, 64, 0><<<dim3(1024), dim3(256), 0, stream>>>(
      xyz, pts16, newxyz, idx0,
      wf16 + WOFF[0], bs[0], wf16 + WOFF[1], bs[1], wf16 + WOFF[2], bs[2], feat);
  // scale 1: K=32, 35->64->64->128, chan offset 64  (Q=4  -> 2048 blocks)
  group_mlp_kernel<32, 64, 64, 128, 64><<<dim3(2048), dim3(256), 0, stream>>>(
      xyz, pts16, newxyz, idx1,
      wf16 + WOFF[3], bs[3], wf16 + WOFF[4], bs[4], wf16 + WOFF[5], bs[5], feat);
  // scale 2: K=64, 35->64->96->128, chan offset 192 (Q=2  -> 4096 blocks)
  group_mlp_kernel<64, 64, 96, 128, 192><<<dim3(4096), dim3(256), 0, stream>>>(
      xyz, pts16, newxyz, idx2,
      wf16 + WOFF[6], bs[6], wf16 + WOFF[7], bs[7], wf16 + WOFF[8], bs[8], feat);

  agg_kernel<<<dim3(64), dim3(256), 0, stream>>>(feat, wf16 + WOFF[9], aggb, newpts);
}
